// CausalAttention_82386062672030
// MI455X (gfx1250) — matrix operand order repaired and checked by reading, not compiled, not run on hardware
//
#include <hip/hip_runtime.h>
#include <hip/hip_bf16.h>

typedef __attribute__((ext_vector_type(16))) _Float16 v16h;
typedef __attribute__((ext_vector_type(8)))  _Float16 v8h;
typedef __attribute__((ext_vector_type(8)))  float    v8f;
typedef __attribute__((ext_vector_type(4))) unsigned int u32x4;
typedef __attribute__((ext_vector_type(8))) int i32x8;
typedef __attribute__((ext_vector_type(4))) int i32x4;

#define Bc  4
#define Tc  2048
#define Cc  1024
#define Hc  16
#define HDc 64

#if defined(__has_builtin)
#if __has_builtin(__builtin_amdgcn_tensor_load_to_lds)
#define HAVE_TDM 1
#endif
#if __has_builtin(__builtin_amdgcn_sched_barrier)
#define SCHED_FENCE() __builtin_amdgcn_sched_barrier(0)
#endif
#endif
#ifndef HAVE_TDM
#define HAVE_TDM 0
#endif
#ifndef SCHED_FENCE
#define SCHED_FENCE() ((void)0)
#endif

#if __has_include(<hip/amd_detail/amd_gfx1250_TDM.h>)
#define TDM_6ARG 1
#else
#define TDM_6ARG 0
#endif

static __device__ __forceinline__ v8f vzero8() {
  v8f z;
#pragma unroll
  for (int i = 0; i < 8; ++i) z[i] = 0.0f;
  return z;
}

static __device__ __forceinline__ v16h make16(v8h lo, v8h hi) {
  v16h r;
#pragma unroll
  for (int i = 0; i < 8; ++i) { r[i] = lo[i]; r[i + 8] = hi[i]; }
  return r;
}

static __device__ __forceinline__ v8f wmma_f16(v16h a, v16h b, v8f c) {
  return __builtin_amdgcn_wmma_f32_16x16x32_f16(false, a, false, b, (short)0, c,
                                                false, false);
}

// Low 32 bits of a flat shared-memory address = LDS byte offset.
static __device__ __forceinline__ unsigned lds_off(const void* p) {
  return (unsigned)(unsigned long)p;
}

#if HAVE_TDM
// TDM: DMA a 2D tile (d0 contiguous f16 x d1 rows) from global into LDS.
// Descriptor per CDNA5 ISA 8.3/8.4: group0 {count=1, lds_addr, global_addr,
// type=2}; group1 {data_size=2B, tensor dims == tile dims (in-bounds),
// dim0 stride in elements}.
static __device__ __forceinline__ void tdm_load_tile(
    const _Float16* g, unsigned ldsoff, long stride_elems,
    unsigned d0, unsigned d1) {
  unsigned long ga = (unsigned long)g;
  u32x4 g0;
  g0[0] = 1u;                                   // count=1, user descriptor
  g0[1] = ldsoff;                               // lds_addr (bytes)
  g0[2] = (unsigned)(ga & 0xffffffffu);         // global_addr[31:0]
  g0[3] = (unsigned)((ga >> 32) & 0x01ffffffu)  // global_addr[56:32]
          | (2u << 30);                         // type=2 ("image")
  i32x8 g1;
  g1[0] = (int)(1u << 16);                      // data_size=1 -> 2 bytes
  g1[1] = (int)(d0 << 16);                      // tensor_dim0[15:0]
  g1[2] = (int)(d1 << 16);                      // tensor_dim1[15:0]
  g1[3] = (int)(d0 << 16);                      // tile_dim0
  g1[4] = (int)d1;                              // tile_dim1 (tile_dim2=0)
  g1[5] = (int)(unsigned)(stride_elems & 0xffffffff);  // dim0_stride[31:0]
  g1[6] = (int)(unsigned)((stride_elems >> 32) & 0xffff);
  g1[7] = 0;
  i32x4 z4 = {0, 0, 0, 0};
#if TDM_6ARG
  i32x8 z8 = {0, 0, 0, 0, 0, 0, 0, 0};
  __builtin_amdgcn_tensor_load_to_lds(g0, g1, z4, z4, z8, 0);
#else
  __builtin_amdgcn_tensor_load_to_lds(g0, g1, z4, z4, 0);
#endif
}
#endif  // HAVE_TDM

// ---------------------------------------------------------------------------
// f32 -> f16 conversion, optionally transposing [rows][cols] -> [cols][rows]
// ---------------------------------------------------------------------------
__global__ void conv_f16_kernel(const float* __restrict__ src,
                                _Float16* __restrict__ dst,
                                long n, int rows, int cols, int transpose) {
  long i = (long)blockIdx.x * blockDim.x + threadIdx.x;
  if (i >= n) return;
  float v = src[i];
  if (transpose) {
    long r = i / cols, c = i % cols;
    dst[c * (long)rows + r] = (_Float16)v;
  } else {
    dst[i] = (_Float16)v;
  }
}

// ---------------------------------------------------------------------------
// WMMA GEMM with TDM-staged, double-buffered 64x64 LDS tiles.
// out[M,N] = A[M,K](f16) * Bt[N,K]^T(f16) + bias
// block = 128 threads (4 waves); block tile 64x64; wave computes 16x64 strip.
// mode 0: f32 row-major to outF
// mode 1: f16 scatter to Q[B,H,T,64]
// mode 2: f16 scatter: cols<1024 -> K[B,H,T,64], cols>=1024 -> Vt[B,H,64,T]
// ---------------------------------------------------------------------------
__global__ void gemm_wmma_kernel(const _Float16* __restrict__ A,
                                 const _Float16* __restrict__ Bt,
                                 const float* __restrict__ bias,
                                 int M, int N, int K, int mode,
                                 float* __restrict__ outF,
                                 _Float16* __restrict__ outQ,
                                 _Float16* __restrict__ outK,
                                 _Float16* __restrict__ outVt) {
  __shared__ alignas(16) _Float16 Atile[2][64][64];
  __shared__ alignas(16) _Float16 Btile[2][64][64];

  const int lane   = threadIdx.x & 31;
  const int wave   = threadIdx.x >> 5;
  const int half16 = lane >> 4;
  const int l16    = lane & 15;
  const long m0blk = (long)blockIdx.y * 64;
  const long n0    = (long)blockIdx.x * 64;

  v8f acc[4];
#pragma unroll
  for (int nb = 0; nb < 4; ++nb) acc[nb] = vzero8();

  // Stage one 64-wide k-step's tiles into LDS buffer `buf`.
  auto stage = [&](int buf, int k) {
#if HAVE_TDM
    if (wave == 0)
      tdm_load_tile(A + m0blk * (long)K + k, lds_off(&Atile[buf][0][0]),
                    (long)K, 64, 64);
    else if (wave == 1)
      tdm_load_tile(Bt + n0 * (long)K + k, lds_off(&Btile[buf][0][0]),
                    (long)K, 64, 64);
#else
    // cooperative fallback: 128 threads copy 2x (64x64) f16 tiles
    const int tid = (int)threadIdx.x;
#pragma unroll
    for (int c = 0; c < 4; ++c) {
      const int chunk = c * 128 + tid;       // 0..511
      const int row = chunk >> 3;
      const int seg = (chunk & 7) * 8;
      *(v8h*)&Atile[buf][row][seg] =
          *(const v8h*)(A + (m0blk + row) * (long)K + k + seg);
      *(v8h*)&Btile[buf][row][seg] =
          *(const v8h*)(Bt + (n0 + row) * (long)K + k + seg);
    }
#endif
  };

  stage(0, 0);
#if HAVE_TDM
  __builtin_amdgcn_s_wait_tensorcnt(0);
#endif
  __syncthreads();

  int cur = 0;
  for (int k = 0; k < K; k += 64) {
    const bool more = (k + 64) < K;
    if (more) stage(cur ^ 1, k + 64);   // DMA next tiles while computing

#pragma unroll
    for (int ks = 0; ks < 2; ++ks) {
      // batch all fragment loads, then issue the 4 WMMAs back-to-back
      const _Float16* ar = &Atile[cur][wave * 16 + l16][ks * 32];
      v16h av = make16(*(const v8h*)(ar + 8 * half16),
                       *(const v8h*)(ar + 16 + 8 * half16));
      v16h bfrag[4];
#pragma unroll
      for (int nb = 0; nb < 4; ++nb) {
        const _Float16* br = &Btile[cur][nb * 16 + l16][ks * 32 + 8 * half16];
        bfrag[nb] = make16(*(const v8h*)br, *(const v8h*)(br + 16));
      }
      SCHED_FENCE();  // keep all DS reads before the WMMA burst
#pragma unroll
      for (int nb = 0; nb < 4; ++nb)
        acc[nb] = wmma_f16(av, bfrag[nb], acc[nb]);
      SCHED_FENCE();
    }

#if HAVE_TDM
    if (more) __builtin_amdgcn_s_wait_tensorcnt(0);
#endif
    __syncthreads();
    cur ^= 1;
  }

#pragma unroll
  for (int nb = 0; nb < 4; ++nb) {
    const long col = n0 + nb * 16 + l16;
    const float bv = bias[col];
#pragma unroll
    for (int r = 0; r < 8; ++r) {
      const long row = m0blk + wave * 16 + r + 8 * half16;
      const float v = acc[nb][r] + bv;
      if (mode == 0) {
        outF[row * (long)N + col] = v;
      } else if (mode == 1) {
        const long b = row / Tc, t = row % Tc;
        const long h = col / HDc, d = col % HDc;
        outQ[(((b * Hc + h) * Tc) + t) * HDc + d] = (_Float16)v;
      } else {
        const long b = row / Tc, t = row % Tc;
        if (col < Cc) {
          const long h = col / HDc, d = col % HDc;
          outK[(((b * Hc + h) * Tc) + t) * HDc + d] = (_Float16)v;
        } else {
          const long c2 = col - Cc;
          const long h = c2 / HDc, d = c2 % HDc;
          outVt[(((b * Hc + h) * HDc) + d) * Tc + t] = (_Float16)v;
        }
      }
    }
  }
}

// ---------------------------------------------------------------------------
// Flash attention with causal mask + ALiBi.
// grid (T/64, H, B), block 128 (4 waves). Each wave owns 16 query rows.
// Q,K: [B,H,T,64] f16.  Vt: [B,H,64,T] f16.  Yh out: [B,T,C] f16.
// ---------------------------------------------------------------------------
__global__ void attn_kernel(const _Float16* __restrict__ Q,
                            const _Float16* __restrict__ Km,
                            const _Float16* __restrict__ Vt,
                            const float* __restrict__ alibi_m,
                            _Float16* __restrict__ Yh) {
  __shared__ alignas(16) _Float16 pLDS[4][16][64];  // per-wave P tile

  const int lane   = threadIdx.x & 31;
  const int wave   = threadIdx.x >> 5;
  const int half16 = lane >> 4;
  const int l16    = lane & 15;
  const int h = blockIdx.y;
  const int b = blockIdx.z;
  const int q0 = blockIdx.x * 64 + wave * 16;  // wave-uniform

  const long bh = (long)b * Hc + h;
  const _Float16* Qrow  = Q  + (bh * Tc + q0 + l16) * HDc;
  const _Float16* Kbase = Km + bh * Tc * (long)HDc;
  const _Float16* Vbase = Vt + bh * (long)HDc * Tc;
  const float mh = alibi_m[h];
  const float NEG = -3.0e38f;

  // Q fragments for the two 32-wide K steps over head dim
  v16h qa[2];
#pragma unroll
  for (int ds = 0; ds < 2; ++ds) {
    v8h lo = *(const v8h*)(Qrow + ds * 32 + 8 * half16);
    v8h hi = *(const v8h*)(Qrow + ds * 32 + 16 + 8 * half16);
    qa[ds] = make16(lo, hi);
  }

  v8f o[4];
#pragma unroll
  for (int nb = 0; nb < 4; ++nb) o[nb] = vzero8();
  float mrow[8], lrow[8];
#pragma unroll
  for (int r = 0; r < 8; ++r) { mrow[r] = NEG; lrow[r] = 0.0f; }

  for (int kt = 0; kt <= q0 + 15; kt += 64) {  // causal bound (wave-uniform)
    // ---- S = Q @ K^T for this 64-key tile ----
    v8f s[4];
#pragma unroll
    for (int nb = 0; nb < 4; ++nb) s[nb] = vzero8();
#pragma unroll
    for (int ds = 0; ds < 2; ++ds) {
      v16h kf[4];
#pragma unroll
      for (int nb = 0; nb < 4; ++nb) {
        const _Float16* kr =
            Kbase + (long)(kt + nb * 16 + l16) * HDc + ds * 32 + 8 * half16;
        kf[nb] = make16(*(const v8h*)kr, *(const v8h*)(kr + 16));
      }
      SCHED_FENCE();  // all K-fragment loads before the WMMA burst
#pragma unroll
      for (int nb = 0; nb < 4; ++nb)
        s[nb] = wmma_f16(qa[ds], kf[nb], s[nb]);
      SCHED_FENCE();
    }

    // ---- scale + ALiBi + causal mask; per-row online softmax stats ----
    float pv[4][8];
    float rmax[8];
#pragma unroll
    for (int r = 0; r < 8; ++r) rmax[r] = NEG;
#pragma unroll
    for (int nb = 0; nb < 4; ++nb) {
      const int kcol = kt + nb * 16 + l16;
#pragma unroll
      for (int r = 0; r < 8; ++r) {
        const int qrow = q0 + r + 8 * half16;
        const float sv = (kcol <= qrow)
                             ? (s[nb][r] * 0.125f + mh * (float)(kcol - qrow))
                             : NEG;
        pv[nb][r] = sv;
        rmax[r] = fmaxf(rmax[r], sv);
      }
    }
#pragma unroll
    for (int off = 8; off >= 1; off >>= 1)
#pragma unroll
      for (int r = 0; r < 8; ++r)
        rmax[r] = fmaxf(rmax[r], __shfl_xor(rmax[r], off, 32));

    float alpha[8], rsum[8];
#pragma unroll
    for (int r = 0; r < 8; ++r) {
      const float mn = fmaxf(mrow[r], rmax[r]);
      alpha[r] = __expf(mrow[r] - mn);
      mrow[r] = mn;
      rsum[r] = 0.0f;
    }
#pragma unroll
    for (int nb = 0; nb < 4; ++nb)
#pragma unroll
      for (int r = 0; r < 8; ++r) {
        pv[nb][r] = __expf(pv[nb][r] - mrow[r]);
        rsum[r] += pv[nb][r];
      }
#pragma unroll
    for (int off = 8; off >= 1; off >>= 1)
#pragma unroll
      for (int r = 0; r < 8; ++r) rsum[r] += __shfl_xor(rsum[r], off, 32);
#pragma unroll
    for (int r = 0; r < 8; ++r) lrow[r] = lrow[r] * alpha[r] + rsum[r];
#pragma unroll
    for (int nb = 0; nb < 4; ++nb)
#pragma unroll
      for (int r = 0; r < 8; ++r) o[nb][r] *= alpha[r];

    // ---- re-layout P (C-layout -> A-layout) through wave-private LDS ----
#pragma unroll
    for (int nb = 0; nb < 4; ++nb)
#pragma unroll
      for (int r = 0; r < 8; ++r)
        pLDS[wave][r + 8 * half16][nb * 16 + l16] = (_Float16)pv[nb][r];
    asm volatile("s_wait_dscnt 0" ::: "memory");

    // ---- O += P @ V ----
#pragma unroll
    for (int ds = 0; ds < 2; ++ds) {
      const _Float16* pr = &pLDS[wave][l16][0];
      v16h pa = make16(*(const v8h*)(pr + ds * 32 + 8 * half16),
                       *(const v8h*)(pr + ds * 32 + 16 + 8 * half16));
      v16h vf[4];
#pragma unroll
      for (int nb = 0; nb < 4; ++nb) {
        const _Float16* vr =
            Vbase + (long)(nb * 16 + l16) * Tc + kt + ds * 32 + 8 * half16;
        vf[nb] = make16(*(const v8h*)vr, *(const v8h*)(vr + 16));
      }
      SCHED_FENCE();  // all P/V fragment loads before the WMMA burst
#pragma unroll
      for (int nb = 0; nb < 4; ++nb)
        o[nb] = wmma_f16(pa, vf[nb], o[nb]);
      SCHED_FENCE();
    }
  }

  // ---- normalize and store Y as f16 [B,T,C] ----
  float inv[8];
#pragma unroll
  for (int r = 0; r < 8; ++r) inv[r] = 1.0f / lrow[r];
#pragma unroll
  for (int nb = 0; nb < 4; ++nb)
#pragma unroll
    for (int r = 0; r < 8; ++r) {
      const int t = q0 + r + 8 * half16;
      const int col = h * HDc + nb * 16 + l16;
      Yh[((long)b * Tc + t) * Cc + col] = (_Float16)(o[nb][r] * inv[r]);
    }
}

// ---------------------------------------------------------------------------
extern "C" void kernel_launch(void* const* d_in, const int* in_sizes, int n_in,
                              void* d_out, int out_size, void* d_ws,
                              size_t ws_size, hipStream_t stream) {
  const float* x       = (const float*)d_in[0];
  const float* Wq      = (const float*)d_in[1];
  const float* bq      = (const float*)d_in[2];
  const float* Wkv     = (const float*)d_in[3];
  const float* bkv     = (const float*)d_in[4];
  const float* Wp      = (const float*)d_in[5];
  const float* bp      = (const float*)d_in[6];
  const float* alibi_m = (const float*)d_in[7];
  (void)in_sizes; (void)n_in; (void)out_size; (void)ws_size;

  char* ws = (char*)d_ws;
  size_t off = 0;
  auto alloc = [&](size_t bytes) {
    char* p = ws + off;
    off += (bytes + 255) & ~(size_t)255;
    return p;
  };
  const long MT = (long)Bc * Tc;  // 8192 rows
  _Float16* Xh   = (_Float16*)alloc((size_t)MT * Cc * 2);
  _Float16* Wqt  = (_Float16*)alloc((size_t)Cc * Cc * 2);
  _Float16* Wkvt = (_Float16*)alloc((size_t)(2 * Cc) * Cc * 2);
  _Float16* Wpt  = (_Float16*)alloc((size_t)Cc * Cc * 2);
  _Float16* Qb   = (_Float16*)alloc((size_t)MT * Cc * 2);
  _Float16* Kb   = (_Float16*)alloc((size_t)MT * Cc * 2);
  _Float16* Vtb  = (_Float16*)alloc((size_t)MT * Cc * 2);
  _Float16* Yh   = (_Float16*)alloc((size_t)MT * Cc * 2);

  // 1) conversions / transposes
  {
    long n = MT * Cc;
    conv_f16_kernel<<<dim3((n + 255) / 256), dim3(256), 0, stream>>>(
        x, Xh, n, 0, 0, 0);
    n = (long)Cc * Cc;
    conv_f16_kernel<<<dim3((n + 255) / 256), dim3(256), 0, stream>>>(
        Wq, Wqt, n, Cc, Cc, 1);
    n = (long)Cc * 2 * Cc;
    conv_f16_kernel<<<dim3((n + 255) / 256), dim3(256), 0, stream>>>(
        Wkv, Wkvt, n, Cc, 2 * Cc, 1);
    n = (long)Cc * Cc;
    conv_f16_kernel<<<dim3((n + 255) / 256), dim3(256), 0, stream>>>(
        Wp, Wpt, n, Cc, Cc, 1);
  }

  // 2) Q = Xh*Wq + bq   (scatter to [B,H,T,64])
  gemm_wmma_kernel<<<dim3(Cc / 64, MT / 64), dim3(128), 0, stream>>>(
      Xh, Wqt, bq, (int)MT, Cc, Cc, 1, nullptr, Qb, nullptr, nullptr);

  // 3) KV = Xh*Wkv + bkv (scatter: K -> [B,H,T,64], V -> Vt [B,H,64,T])
  gemm_wmma_kernel<<<dim3(2 * Cc / 64, MT / 64), dim3(128), 0, stream>>>(
      Xh, Wkvt, bkv, (int)MT, 2 * Cc, Cc, 2, nullptr, nullptr, Kb, Vtb);

  // 4) flash attention with causal + ALiBi
  attn_kernel<<<dim3(Tc / 64, Hc, Bc), dim3(128), 0, stream>>>(
      Qb, Kb, Vtb, alibi_m, Yh);

  // 5) out = Yh*Wp + bp (f32)
  gemm_wmma_kernel<<<dim3(Cc / 64, MT / 64), dim3(128), 0, stream>>>(
      Yh, Wpt, bp, (int)MT, Cc, Cc, 0, (float*)d_out, nullptr, nullptr, nullptr);
}
